// Bandau_Pointer_75187697483954
// MI455X (gfx1250) — matrix operand order repaired and checked by reading, not compiled, not run on hardware
//
#include <hip/hip_runtime.h>
#include <hip/hip_bf16.h>
#include <math.h>

typedef __bf16 v16bf __attribute__((ext_vector_type(16)));
typedef __bf16 v8bf  __attribute__((ext_vector_type(8)));
typedef __bf16 v4bf  __attribute__((ext_vector_type(4)));
typedef float  v8f   __attribute__((ext_vector_type(8)));

#define D_MODEL 1024
#define N_CITY  2048
#define BATCH   32
#define M_TILE  64
#define XS_STRIDE 1032   // 1024 + 8 bf16 pad -> row stride 2064B, 16B aligned, bank-spread

// CDNA5 has V_TANH_F32 (trans op). Use the hardware instruction when the
// builtin exists; otherwise a branch-free rational approx (never libm's
// branchy exp expansion).
__device__ __forceinline__ float fast_tanh(float x) {
#if __has_builtin(__builtin_amdgcn_tanhf)
    return __builtin_amdgcn_tanhf(x);
#elif __has_builtin(__builtin_amdgcn_tanh_f32)
    return __builtin_amdgcn_tanh_f32(x);
#else
    // tanh(x) ~= x*(27+x^2)/(27+9x^2) clamped; branch-free
    float cx = fminf(fmaxf(x, -9.0f), 9.0f);
    float x2 = cx * cx;
    return cx * (27.0f + x2) / (27.0f + 9.0f * x2);
#endif
}

// ---------------- Kernel 1: W1 fp32 -> bf16 (workspace, L2-resident 2MB) ----------------
__global__ void w1_to_bf16_kernel(const float* __restrict__ W1, __bf16* __restrict__ W1b) {
    int i = (blockIdx.x * 256 + threadIdx.x) * 4;
    float4 f = *(const float4*)(W1 + i);
    v4bf o;
    o[0] = (__bf16)f.x; o[1] = (__bf16)f.y; o[2] = (__bf16)f.z; o[3] = (__bf16)f.w;
    *(v4bf*)(W1b + i) = o;
}

// ---------------- Kernel 2: query[b,e] = x[b, N, :] . W2[e, :]  (tiny, fp32) ----------------
__global__ void query_kernel(const float* __restrict__ x, const float* __restrict__ W2,
                             float* __restrict__ qbuf) {
    __shared__ float xl[D_MODEL];
    const int b = blockIdx.x, tid = threadIdx.x;
    const float* xr = x + ((size_t)b * (N_CITY + 1) + N_CITY) * D_MODEL;
    *(float4*)(xl + tid * 4) = *(const float4*)(xr + tid * 4);
    __syncthreads();
    for (int e = tid; e < D_MODEL; e += 256) {
        const float* w = W2 + (size_t)e * D_MODEL;
        float acc = 0.f;
        for (int d = 0; d < D_MODEL; d += 4) {
            float4 wv = *(const float4*)(w + d);
            acc += wv.x * xl[d] + wv.y * xl[d + 1] + wv.z * xl[d + 2] + wv.w * xl[d + 3];
        }
        qbuf[(size_t)b * D_MODEL + e] = acc;
    }
}

// ---------------- Main kernel: bf16 WMMA GEMM + fused tanh/dot epilogue ----------------
__launch_bounds__(256)
__global__ void energy_kernel(const float* __restrict__ x, const __bf16* __restrict__ W1b,
                              const float* __restrict__ qbuf, const float* __restrict__ V,
                              float* __restrict__ out) {
    extern __shared__ char smem[];
    __bf16* xs   = (__bf16*)smem;                                        // 64 x 1032 bf16
    float*  elds = (float*)(smem + M_TILE * XS_STRIDE * sizeof(__bf16)); // 64 energies

    const int tid    = threadIdx.x;
    const int lane   = tid & 31;
    const int wave   = tid >> 5;         // 0..7
    const int mhalf  = wave & 1;         // 32-row half
    const int equart = wave >> 1;        // 64-col quarter of a 256-wide e tile
    const int lhalf  = lane >> 4;        // 0/1
    const int l15    = lane & 15;

    const int mrow0 = blockIdx.x * M_TILE;       // flat row in [B*N]
    const int b     = mrow0 >> 11;               // / 2048 (64 | 2048 -> uniform per block)
    const int n0    = mrow0 & (N_CITY - 1);

    // ---- Stage x tile (64 rows x 1024 K) into LDS as bf16, coalesced ----
    {
        const float* xb = x + ((size_t)b * (N_CITY + 1) + n0) * D_MODEL;
        for (int j = 0; j < M_TILE; ++j) {
            float4 f = *(const float4*)(xb + (size_t)j * D_MODEL + tid * 4);
            v4bf o;
            o[0] = (__bf16)f.x; o[1] = (__bf16)f.y; o[2] = (__bf16)f.z; o[3] = (__bf16)f.w;
            *(v4bf*)(xs + j * XS_STRIDE + tid * 4) = o;
        }
    }
    if (tid < M_TILE) elds[tid] = 0.f;
    __syncthreads();

    const float* qrow = qbuf + (size_t)b * D_MODEL;

    for (int eiter = 0; eiter < 4; ++eiter) {
        const int ebase = eiter * 256 + equart * 64;

        float qv[4], vv[4];
#pragma unroll
        for (int s = 0; s < 4; ++s) {
            int e = ebase + s * 16 + l15;
            qv[s] = qrow[e];
            vv[s] = V[e];
        }

        v8f acc[2][4];
#pragma unroll
        for (int f = 0; f < 2; ++f)
#pragma unroll
            for (int s = 0; s < 4; ++s) acc[f][s] = (v8f)0.f;

#pragma unroll 2
        for (int k = 0; k < D_MODEL; k += 32) {
            // A fragments (16x32 bf16): lane<16 holds K {0..15}, lane>=16 {16..31}
            v16bf a[2];
#pragma unroll
            for (int f = 0; f < 2; ++f) {
                const int row = mhalf * 32 + f * 16 + l15;
                const int klo = k + lhalf * 16;
                const __bf16* base = xs + row * XS_STRIDE + klo;
                union { v16bf v; v8bf h[2]; } u;
                u.h[0] = *(const v8bf*)(base);
                u.h[1] = *(const v8bf*)(base + 8);
                a[f] = u.v;
            }
            // B fragments (32x16 bf16): lane n holds 16 contiguous K of W1 row n
            v16bf bm[4];
#pragma unroll
            for (int s = 0; s < 4; ++s) {
                const int ncol = ebase + s * 16 + l15;
                const int kk   = k + lhalf * 16;
                bm[s] = *(const v16bf*)(W1b + (size_t)ncol * D_MODEL + kk);
            }
#pragma unroll
            for (int f = 0; f < 2; ++f)
#pragma unroll
                for (int s = 0; s < 4; ++s)
                    acc[f][s] = __builtin_amdgcn_wmma_f32_16x16x32_bf16(
                        false, a[f], false, bm[s], (short)0, acc[f][s], false, false);
        }

        // ---- Fused epilogue: tanh(keys + q) * V, reduce over e ----
#pragma unroll
        for (int f = 0; f < 2; ++f) {
            float part[8];
#pragma unroll
            for (int r = 0; r < 8; ++r) part[r] = 0.f;
#pragma unroll
            for (int s = 0; s < 4; ++s)
#pragma unroll
                for (int r = 0; r < 8; ++r)
                    part[r] += fast_tanh(acc[f][s][r] + qv[s]) * vv[s];
#pragma unroll
            for (int r = 0; r < 8; ++r) {
                float v = part[r];
                v += __shfl_xor(v, 1, 32);
                v += __shfl_xor(v, 2, 32);
                v += __shfl_xor(v, 4, 32);
                v += __shfl_xor(v, 8, 32);   // sum over 16 N-lanes
                if (l15 == 0)
                    atomicAdd(&elds[mhalf * 32 + f * 16 + r + 8 * lhalf], v);
            }
        }
    }

    __syncthreads();
    if (tid < M_TILE) out[mrow0 + tid] = elds[tid];   // out is flat [B*N]
}

extern "C" void kernel_launch(void* const* d_in, const int* in_sizes, int n_in,
                              void* d_out, int out_size, void* d_ws, size_t ws_size,
                              hipStream_t stream) {
    const float* x  = (const float*)d_in[0];   // [32, 2049, 1024] fp32
    const float* W1 = (const float*)d_in[1];   // [1024, 1024] fp32
    const float* W2 = (const float*)d_in[2];   // [1024, 1024] fp32
    const float* V  = (const float*)d_in[3];   // [1024] fp32
    float* out = (float*)d_out;                // [32, 1, 2048] fp32

    __bf16* W1b = (__bf16*)d_ws;                                           // 2 MB
    float*  qb  = (float*)((char*)d_ws + (size_t)D_MODEL * D_MODEL * 2);   // 128 KB

    w1_to_bf16_kernel<<<(D_MODEL * D_MODEL) / (256 * 4), 256, 0, stream>>>(W1, W1b);
    query_kernel<<<BATCH, 256, 0, stream>>>(x, W2, qb);

    const size_t shmem = (size_t)M_TILE * XS_STRIDE * sizeof(__bf16) + M_TILE * sizeof(float);
    hipFuncSetAttribute(reinterpret_cast<const void*>(energy_kernel),
                        hipFuncAttributeMaxDynamicSharedMemorySize, (int)shmem);
    energy_kernel<<<(BATCH * N_CITY) / M_TILE, 256, shmem, stream>>>(x, W1b, qb, V, out);
}
